// BiMamba_43739946942604
// MI455X (gfx1250) — compile-verified, matrix operand order checked by reading, not run on hardware
//
#include <hip/hip_runtime.h>
#include <math.h>

// ---- Problem constants (match reference) ----
#define D_MODEL  512
#define D_STATE  16
#define D_CONVK  4
#define D_INNER  1024
#define DT_RANK  32
#define BATCH    4
#define SEQ      2048
#define NBLK     4
#define MROWS    (BATCH * SEQ)   // 8192

typedef __attribute__((ext_vector_type(16))) _Float16 v16h;
typedef __attribute__((ext_vector_type(8)))  _Float16 v8h;
typedef __attribute__((ext_vector_type(8)))  float    v8f;

__device__ __forceinline__ float sigmoidf_(float x) { return 1.0f / (1.0f + __expf(-x)); }
__device__ __forceinline__ float siluf_(float x)    { return x * sigmoidf_(x); }
__device__ __forceinline__ float softplusf_(float x){ return (x > 20.0f) ? x : log1pf(__expf(x)); }

// =====================================================================
// One-time pre-pass: WT[n*K + k] = (f16) W[k*N + n]
// LDS-tiled 32x32 so both global sides stay coalesced.
// =====================================================================
__global__ __launch_bounds__(256) void transpose_cvt_kernel(
    const float* __restrict__ W, _Float16* __restrict__ WT, int K, int N)
{
    __shared__ float tile[32][33];
    const int tx = threadIdx.x & 31;
    const int ty = threadIdx.x >> 5;        // 0..7
    const int kt = blockIdx.x * 32;
    const int nt = blockIdx.y * 32;
    #pragma unroll
    for (int j = 0; j < 4; ++j)
        tile[ty + 8 * j][tx] = W[(size_t)(kt + ty + 8 * j) * N + nt + tx];
    __syncthreads();
    #pragma unroll
    for (int j = 0; j < 4; ++j)
        WT[(size_t)(nt + ty + 8 * j) * K + kt + tx] = (_Float16)tile[tx][ty + 8 * j];
}

// Elementwise f32 -> f16
__global__ __launch_bounds__(256) void cvt_f16_kernel(
    const float* __restrict__ in, _Float16* __restrict__ out, int n)
{
    int i = blockIdx.x * 256 + threadIdx.x;
    if (i < n) out[i] = (_Float16)in[i];
}

// =====================================================================
// GEMM: C[M,N] = act(A[M,K] @ W[K,N] + bias), f16 operands in memory,
// f32 accumulate via v_wmma_f32_16x16x32_f16. W pre-transposed (N x K).
// Block = 128 threads / 4 waves; block tile 64x64, wave tile 16x64.
// Fragments loaded directly from global (L2-resident) as b128 pairs per
// the ISA 16-bit layout (lane&15 -> row/col, lane>>4 -> K-half).
// Register double-buffering: iteration k prefetches k+1's fragments
// before issuing its 4 WMMAs, so WMMAs never wait on just-issued loads.
// =====================================================================
template <int ACT, bool FLIPA, bool FLIPC, bool HASBIAS, bool STF32, bool STF16,
          int LDA, int LDC, int K>
__global__ __launch_bounds__(128) void gemm_wmma(
    const _Float16* __restrict__ A,
    const _Float16* __restrict__ WT,
    const float* __restrict__ bias,
    float* __restrict__ Cf, _Float16* __restrict__ Ch,
    int colOff)
{
    const int tid  = threadIdx.x;
    const int wave = tid >> 5;
    const int lane = tid & 31;
    const int tileN = blockIdx.y * 64;
    const int waveM = blockIdx.x * 64 + wave * 16;
    const int r  = lane & 15;          // row (A) / col (B) within fragment
    const int kh = (lane >> 4) * 8;    // lane-dependent K offset

    int rowA = waveM + r;
    if (FLIPA) rowA = (rowA & ~(SEQ - 1)) + (SEQ - 1 - (rowA & (SEQ - 1)));
    const _Float16* pa = A + (size_t)rowA * LDA + kh;

    const _Float16* pb[4];
    #pragma unroll
    for (int t = 0; t < 4; ++t)
        pb[t] = WT + (size_t)(tileN + t * 16 + r) * K + kh;

    v8f acc[4];
    #pragma unroll
    for (int t = 0; t < 4; ++t)
        #pragma unroll
        for (int e = 0; e < 8; ++e) acc[t][e] = 0.0f;

    union Frag { v16h v; v8h h[2]; };

    // preload k = 0 fragments
    Frag ua, ub[4];
    ua.h[0] = *(const v8h*)(pa);
    ua.h[1] = *(const v8h*)(pa + 16);
    #pragma unroll
    for (int t = 0; t < 4; ++t) {
        ub[t].h[0] = *(const v8h*)(pb[t]);
        ub[t].h[1] = *(const v8h*)(pb[t] + 16);
    }

    #pragma unroll 4
    for (int k0 = 0; k0 < K; k0 += 32) {
        const int kn = k0 + 32;
        Frag na, nb[4];
        if (kn < K) {   // prefetch next iteration's fragments
            na.h[0] = *(const v8h*)(pa + kn);
            na.h[1] = *(const v8h*)(pa + kn + 16);
            #pragma unroll
            for (int t = 0; t < 4; ++t) {
                nb[t].h[0] = *(const v8h*)(pb[t] + kn);
                nb[t].h[1] = *(const v8h*)(pb[t] + kn + 16);
            }
        }
        #pragma unroll
        for (int t = 0; t < 4; ++t)
            acc[t] = __builtin_amdgcn_wmma_f32_16x16x32_f16(
                false, ua.v, false, ub[t].v, (short)0, acc[t], false, false);
        if (kn < K) {
            ua = na;
            #pragma unroll
            for (int t = 0; t < 4; ++t) ub[t] = nb[t];
        }
    }

    // Epilogue. D layout: VGPR e of lane -> (M = e + 8*(lane>>4), N = lane&15)
    float bv[4];
    if (HASBIAS) {
        #pragma unroll
        for (int t = 0; t < 4; ++t) bv[t] = bias[tileN + t * 16 + r];
    }
    const int rb = 8 * (lane >> 4);
    #pragma unroll
    for (int e = 0; e < 8; ++e) {
        int gm = waveM + rb + e;
        int drow = gm;
        if (FLIPC) drow = (gm & ~(SEQ - 1)) + (SEQ - 1 - (gm & (SEQ - 1)));
        const size_t rbase = (size_t)drow * LDC + colOff;
        #pragma unroll
        for (int t = 0; t < 4; ++t) {
            int gcol = tileN + t * 16 + r;
            float v = acc[t][e];
            if (HASBIAS) v += bv[t];
            if (ACT == 1) v = softplusf_(v);
            if (STF32) Cf[rbase + gcol] = v;
            if (STF16) Ch[rbase + gcol] = (_Float16)v;
        }
    }
}

// =====================================================================
// Depthwise causal conv (width 4) + bias + SiLU over the xc half of XZ.
// Writes f32 (for the scan) and f16 (A-operand of the x_proj GEMM).
// =====================================================================
__global__ __launch_bounds__(256) void conv_silu_kernel(
    const float* __restrict__ XZ,
    const float* __restrict__ cw,   // (D_INNER, 4)
    const float* __restrict__ cb,   // (D_INNER)
    float* __restrict__ XC,
    _Float16* __restrict__ XCh)
{
    int idx = blockIdx.x * 256 + threadIdx.x;     // over MROWS * D_INNER
    int d = idx & (D_INNER - 1);
    int m = idx >> 10;
    int l = m & (SEQ - 1);
    float acc = cb[d];
    #pragma unroll
    for (int j = 0; j < D_CONVK; ++j) {
        int ll = l - (D_CONVK - 1) + j;
        if (ll >= 0)
            acc += cw[d * D_CONVK + j] * XZ[((size_t)(m - l + ll) << 11) + d];
    }
    float v = siluf_(acc);
    XC[(size_t)m * D_INNER + d]  = v;
    XCh[(size_t)m * D_INNER + d] = (_Float16)v;
}

// =====================================================================
// Selective scan (sequential over L) fused with +x*D and *silu(z) gating.
// One thread per (batch, channel): h[16], A[16] in VGPRs.
// B_t / C_t have a block-uniform address -> scalar/broadcast loads; no
// LDS and no barriers on the latency-critical serial path.
// =====================================================================
__global__ __launch_bounds__(256) void scan_kernel(
    const float* __restrict__ DT,    // (MROWS, D_INNER)
    const float* __restrict__ XC,    // (MROWS, D_INNER)
    const float* __restrict__ XDBL,  // (MROWS, 64): [.,32:48]=B, [.,48:64]=C
    const float* __restrict__ XZ,    // (MROWS, 2048): z at cols [1024,2048)
    const float* __restrict__ A_log, // (D_INNER, 16)
    const float* __restrict__ Dv,    // (D_INNER)
    _Float16* __restrict__ Yh)       // (MROWS, D_INNER)
{
    const int b = blockIdx.x >> 2;
    const int d = ((blockIdx.x & 3) << 8) + threadIdx.x;

    float Areg[D_STATE], h[D_STATE];
    #pragma unroll
    for (int s = 0; s < D_STATE; ++s) {
        Areg[s] = -__expf(A_log[(size_t)d * D_STATE + s]);
        h[s] = 0.0f;
    }
    const float Dd = Dv[d];

    for (int t = 0; t < SEQ; ++t) {
        const size_t row = (size_t)b * SEQ + t;

        // block-uniform B_t / C_t (vectorized, broadcast from cache)
        float Bv[D_STATE], Cv[D_STATE];
        #pragma unroll
        for (int s = 0; s < D_STATE; s += 4) {
            float4 vb = *(const float4*)(XDBL + row * 64 + DT_RANK + s);
            Bv[s] = vb.x; Bv[s + 1] = vb.y; Bv[s + 2] = vb.z; Bv[s + 3] = vb.w;
            float4 vc = *(const float4*)(XDBL + row * 64 + DT_RANK + D_STATE + s);
            Cv[s] = vc.x; Cv[s + 1] = vc.y; Cv[s + 2] = vc.z; Cv[s + 3] = vc.w;
        }

        const float dt  = DT[row * D_INNER + d];
        const float x   = XC[row * D_INNER + d];
        const float dtx = dt * x;
        float y = 0.0f;
        #pragma unroll
        for (int s = 0; s < D_STATE; ++s) {
            h[s] = h[s] * __expf(dt * Areg[s]) + dtx * Bv[s];
            y += h[s] * Cv[s];
        }
        const float z = XZ[row * (2 * D_INNER) + D_INNER + d];
        Yh[row * D_INNER + d] = (_Float16)((y + x * Dd) * siluf_(z));
    }
}

// =====================================================================
// Host-side orchestration (graph-capture safe: only kernel launches)
// =====================================================================
extern "C" void kernel_launch(void* const* d_in, const int* in_sizes, int n_in,
                              void* d_out, int out_size, void* d_ws, size_t ws_size,
                              hipStream_t stream)
{
    (void)in_sizes; (void)n_in; (void)out_size; (void)ws_size;

    const float* x        = (const float*)d_in[0];
    const float* in_proj  = (const float*)d_in[1];  // (4, 512, 2048)
    const float* conv_w   = (const float*)d_in[2];  // (4, 1024, 4)
    const float* conv_b   = (const float*)d_in[3];  // (4, 1024)
    const float* x_proj   = (const float*)d_in[4];  // (4, 1024, 64)
    const float* dt_proj  = (const float*)d_in[5];  // (4, 32, 1024)
    const float* dt_b     = (const float*)d_in[6];  // (4, 1024)
    const float* A_log    = (const float*)d_in[7];  // (4, 1024, 16)
    const float* D_param  = (const float*)d_in[8];  // (4, 1024)
    const float* out_proj = (const float*)d_in[9];  // (4, 1024, 512)
    float* out = (float*)d_out;                     // (4, 2048, 1024)

    // ---- workspace layout ----
    float* XZ   = (float*)d_ws;                         // 8192*2048 f32
    float* XC   = XZ   + (size_t)MROWS * 2 * D_INNER;   // 8192*1024 f32
    float* XDBL = XC   + (size_t)MROWS * D_INNER;       // 8192*64   f32
    float* DT   = XDBL + (size_t)MROWS * 64;            // 8192*1024 f32
    _Float16* Xh    = (_Float16*)(DT + (size_t)MROWS * D_INNER);
    _Float16* XCh   = Xh    + (size_t)MROWS * D_MODEL;          // 8192*1024
    _Float16* XDBLh = XCh   + (size_t)MROWS * D_INNER;          // 8192*64
    _Float16* Yh    = XDBLh + (size_t)MROWS * 64;               // 8192*1024
    _Float16* HBh   = Yh    + (size_t)MROWS * D_INNER;          // 8192*512
    _Float16* WiT   = HBh   + (size_t)MROWS * D_MODEL;          // 4*2048*512
    _Float16* WxT   = WiT   + (size_t)NBLK * 2 * D_INNER * D_MODEL;   // 4*64*1024
    _Float16* WdtT  = WxT   + (size_t)NBLK * 64 * D_INNER;            // 4*1024*32
    _Float16* WoT   = WdtT  + (size_t)NBLK * D_INNER * DT_RANK;       // 4*512*1024

    // ---- pre-pass: transpose+convert all weights to f16; convert x ----
    for (int blk = 0; blk < NBLK; ++blk) {
        transpose_cvt_kernel<<<dim3(D_MODEL / 32, (2 * D_INNER) / 32), 256, 0, stream>>>(
            in_proj + (size_t)blk * D_MODEL * 2 * D_INNER,
            WiT + (size_t)blk * 2 * D_INNER * D_MODEL, D_MODEL, 2 * D_INNER);
        transpose_cvt_kernel<<<dim3(D_INNER / 32, 64 / 32), 256, 0, stream>>>(
            x_proj + (size_t)blk * D_INNER * 64,
            WxT + (size_t)blk * 64 * D_INNER, D_INNER, 64);
        transpose_cvt_kernel<<<dim3(DT_RANK / 32, D_INNER / 32), 256, 0, stream>>>(
            dt_proj + (size_t)blk * DT_RANK * D_INNER,
            WdtT + (size_t)blk * D_INNER * DT_RANK, DT_RANK, D_INNER);
        transpose_cvt_kernel<<<dim3(D_INNER / 32, D_MODEL / 32), 256, 0, stream>>>(
            out_proj + (size_t)blk * D_INNER * D_MODEL,
            WoT + (size_t)blk * D_MODEL * D_INNER, D_INNER, D_MODEL);
    }
    cvt_f16_kernel<<<(MROWS * D_MODEL) / 256, 256, 0, stream>>>(x, Xh, MROWS * D_MODEL);

    const dim3 gXZ(MROWS / 64, (2 * D_INNER) / 64);   // in_proj
    const dim3 gXD(MROWS / 64, 1);                    // x_proj (N=64)
    const dim3 gDT(MROWS / 64, D_INNER / 64);         // dt_proj
    const dim3 gOUT(MROWS / 64, D_MODEL / 64);        // out_proj

    for (int blk = 0; blk < NBLK; ++blk) {
        const bool firstLayer = (blk == 0 || blk == 2);
        const bool lastLayer  = (blk == 1 || blk == 3);
        const _Float16* S = firstLayer ? Xh : HBh;

        const float*    cw   = conv_w + (size_t)blk * D_INNER * D_CONVK;
        const float*    cb   = conv_b + (size_t)blk * D_INNER;
        const float*    bdt  = dt_b   + (size_t)blk * D_INNER;
        const float*    Al   = A_log  + (size_t)blk * D_INNER * D_STATE;
        const float*    Dv   = D_param+ (size_t)blk * D_INNER;
        const _Float16* WiTb = WiT  + (size_t)blk * 2 * D_INNER * D_MODEL;
        const _Float16* WxTb = WxT  + (size_t)blk * 64 * D_INNER;
        const _Float16* WdtTb= WdtT + (size_t)blk * D_INNER * DT_RANK;
        const _Float16* WoTb = WoT  + (size_t)blk * D_MODEL * D_INNER;

        // 1) XZ = S @ Wi  (block 2 reads x time-flipped)
        if (blk == 2)
            gemm_wmma<0, true,  false, false, true, false, D_MODEL, 2 * D_INNER, D_MODEL>
                <<<gXZ, 128, 0, stream>>>(S, WiTb, nullptr, XZ, nullptr, 0);
        else
            gemm_wmma<0, false, false, false, true, false, D_MODEL, 2 * D_INNER, D_MODEL>
                <<<gXZ, 128, 0, stream>>>(S, WiTb, nullptr, XZ, nullptr, 0);

        // 2) XC(+XCh) = silu(causal_depthwise_conv(XZ[:, :1024]) + cb)
        conv_silu_kernel<<<(MROWS * D_INNER) / 256, 256, 0, stream>>>(XZ, cw, cb, XC, XCh);

        // 3) XDBL(+f16) = XC @ Wx     (8192x1024 @ 1024x64)
        gemm_wmma<0, false, false, false, true, true, D_INNER, 64, D_INNER>
            <<<gXD, 128, 0, stream>>>(XCh, WxTb, nullptr, XDBL, XDBLh, 0);

        // 4) DT = softplus(XDBL[:, :32] @ Wdt + bdt)
        gemm_wmma<1, false, false, true, true, false, 64, D_INNER, DT_RANK>
            <<<gDT, 128, 0, stream>>>(XDBLh, WdtTb, bdt, DT, nullptr, 0);

        // 5) selective scan + gating -> Yh (f16)
        scan_kernel<<<BATCH * 4, 256, 0, stream>>>(DT, XC, XDBL, XZ, Al, Dv, Yh);

        // 6) output projection
        if (lastLayer) {
            if (blk == 3)   // un-flip on store into second half of output
                gemm_wmma<0, false, true,  false, true, false, D_INNER, 2 * D_MODEL, D_INNER>
                    <<<gOUT, 128, 0, stream>>>(Yh, WoTb, nullptr, out, nullptr, D_MODEL);
            else            // forward result into first half
                gemm_wmma<0, false, false, false, true, false, D_INNER, 2 * D_MODEL, D_INNER>
                    <<<gOUT, 128, 0, stream>>>(Yh, WoTb, nullptr, out, nullptr, 0);
        } else {
            gemm_wmma<0, false, false, false, false, true, D_INNER, D_MODEL, D_INNER>
                <<<gOUT, 128, 0, stream>>>(Yh, WoTb, nullptr, nullptr, HBh, 0);
        }
    }
}
